// MultiHeadSliddingWindowAttention_37976100831414
// MI455X (gfx1250) — matrix operand order repaired and checked by reading, not compiled, not run on hardware
//
#include <hip/hip_runtime.h>
#include <hip/hip_bf16.h>

typedef _Float16 h16;
typedef __attribute__((ext_vector_type(16))) _Float16 v16h;
typedef __attribute__((ext_vector_type(8)))  float    v8f;

#define EMBED   768
#define NH      12
#define DHEAD   64
#define NSEQ    2048
#define BB      2
#define MROWS   (BB * NSEQ)        // 4096
#define NDIAG   257
#define KPAD    288                // 257 padded to multiple of 32
#define TAILN   (NSEQ - NDIAG)     // 1791

// ---------------------------------------------------------------------------
// WMMA fragment loaders (wave32, 16x16x32 f16 shapes; contiguous k order, same for A and B)
// A 16x32: lane L holds row (L&15); elem e = K[k0 + (L>=16?16:0) + e]
// B 32x16: lane L holds col (L&15); elem e = K[k0 + (L>=16?16:0) + e]
// C/D 16x16 f32: VGPR v, lane L -> row v + (L>=16?8:0), col (L&15)
// ---------------------------------------------------------------------------
__device__ __forceinline__ v16h load_a16x32(const h16* tile, int ld, int k0) {
  int lane = threadIdx.x & 31;
  int lo = lane & 15, hi = lane >> 4;
  const h16* q = tile + (size_t)lo * ld + k0 + hi * 16;
  v16h a;
#pragma unroll
  for (int e = 0; e < 8; ++e) a[e] = q[e];
#pragma unroll
  for (int e = 0; e < 8; ++e) a[e + 8] = q[8 + e];
  return a;
}

// B fragment from "transposed" storage: bt[col][k], row stride = ldk
__device__ __forceinline__ v16h load_b32x16T(const h16* bt, int ldk, int c0, int k0) {
  int lane = threadIdx.x & 31;
  int lo = lane & 15, hi = lane >> 4;
  const h16* q = bt + (size_t)(c0 + lo) * ldk + k0 + hi * 16;
  v16h b;
#pragma unroll
  for (int e = 0; e < 16; ++e) b[e] = q[e];
  return b;
}

__device__ __forceinline__ v8f wmma_f16(v16h a, v16h b, v8f c) {
  return __builtin_amdgcn_wmma_f32_16x16x32_f16(false, a, false, b, (short)0, c,
                                                false, false);
}

// Async global->LDS copy of one 16-byte chunk per lane (CDNA5 ASYNCcnt path)
__device__ __forceinline__ void async_b128_to_lds(unsigned lds_byte_addr,
                                                  const void* gaddr) {
  asm volatile("global_load_async_to_lds_b128 %0, %1, off"
               :: "v"(lds_byte_addr), "v"(gaddr)
               : "memory");
}
__device__ __forceinline__ void wait_asynccnt0() {
  asm volatile("s_wait_asynccnt 0" ::: "memory");
}

// ---------------------------------------------------------------------------
// K0a: f32 -> f16 elementwise
// ---------------------------------------------------------------------------
__global__ void k_cvt_f16(const float* __restrict__ src, h16* __restrict__ dst, int n) {
  int i = blockIdx.x * blockDim.x + threadIdx.x;
  if (i < n) dst[i] = (h16)src[i];
}

// K0b: f32 (K x C row-major) -> f16 transposed (C x K)
__global__ void k_cvt_T(const float* __restrict__ src, h16* __restrict__ dst, int K, int C) {
  int i = blockIdx.x * blockDim.x + threadIdx.x;
  if (i < K * C) {
    int k = i / C, c = i % C;
    dst[(size_t)c * K + k] = (h16)src[i];
  }
}

// ---------------------------------------------------------------------------
// K1/K4: WMMA GEMM  Y(4096x768) = A @ Bt^T + bias
//  Block = 128 threads (4 waves). The 16x768 f16 A strip (24KB) is staged into
//  LDS once via GLOBAL_LOAD_ASYNC_TO_LDS_B128 and shared by all 4 waves.
//  Each wave computes a 16x64 output tile: one LDS A-frag reused across 4
//  WMMAs per k-step (global load instrs per WMMA cut ~2.5x vs 16x16 tiling).
//   mode 0: scatter f16 to (b,h,n,d) layout (QKV projections)
//   mode 1: f32 row-major (final output)
//  grid = (MROWS/16, 768/256)
// ---------------------------------------------------------------------------
__global__ __launch_bounds__(128) void k_gemm16(const h16* __restrict__ A,
                                                const h16* __restrict__ Bt,
                                                const float* __restrict__ bias,
                                                void* __restrict__ outp,
                                                int mode) {
  const int Kdim = EMBED;
  int mt = blockIdx.x;
  int tid = threadIdx.x;
  int wave = tid >> 5;
  int lane = tid & 31;
  int lo = lane & 15, hi = lane >> 4;

  __shared__ __align__(16) h16 As[16 * EMBED];   // 24 KB

  // --- stage A strip: 24KB = 1536 x b128, 12 chunks per thread, async ---
  {
    const char* gsrc = (const char*)(A + (size_t)mt * 16 * Kdim);
    unsigned lbase = (unsigned)(uintptr_t)&As[0];
#pragma unroll
    for (int i = 0; i < 12; ++i) {
      unsigned byteoff = (unsigned)(tid + i * 128) * 16u;
      async_b128_to_lds(lbase + byteoff, gsrc + byteoff);
    }
    wait_asynccnt0();
  }
  __syncthreads();

  int ctbase = blockIdx.y * 16 + wave * 4;       // first of 4 col-tiles

  v8f acc[4];
#pragma unroll
  for (int t = 0; t < 4; ++t) {
    float bv = bias[(ctbase + t) * 16 + lo];
#pragma unroll
    for (int v = 0; v < 8; ++v) acc[t][v] = bv;
  }

  for (int k0 = 0; k0 < Kdim; k0 += 32) {
    if (k0 + 32 < Kdim)
      __builtin_prefetch(Bt + (size_t)(ctbase * 16 + lo) * Kdim + k0 + 32, 0, 0);
    v16h a = load_a16x32(As, Kdim, k0);          // ds_load_b128 x2
#pragma unroll
    for (int t = 0; t < 4; ++t) {
      v16h b = load_b32x16T(Bt, Kdim, (ctbase + t) * 16, k0);
      acc[t] = wmma_f16(a, b, acc[t]);
    }
  }

  if (mode == 0) {
    h16* dst = (h16*)outp;
#pragma unroll
    for (int t = 0; t < 4; ++t) {
#pragma unroll
      for (int v = 0; v < 8; ++v) {
        int row = mt * 16 + v + hi * 8;          // global token row in [0,4096)
        int col = (ctbase + t) * 16 + lo;        // in [0,768)
        int bb = row >> 11, n = row & (NSEQ - 1);
        int hh = col >> 6, d = col & (DHEAD - 1);
        dst[(((size_t)(bb * NH + hh) * NSEQ) + n) * DHEAD + d] = (h16)acc[t][v];
      }
    }
  } else {
    float* dst = (float*)outp;
#pragma unroll
    for (int t = 0; t < 4; ++t) {
#pragma unroll
      for (int v = 0; v < 8; ++v)
        dst[(size_t)(mt * 16 + v + hi * 8) * EMBED + (ctbase + t) * 16 + lo] =
            acc[t][v];
    }
  }
}

// ---------------------------------------------------------------------------
// K3: per (b,h): Vtail[d] = sum_{j=257..2047} V[j][d];
//      Vth[d][j] = V[j][d] for j < 288 (transposed head of V)
// grid = 24 blocks, 64 threads (one per d)
// ---------------------------------------------------------------------------
__global__ void k_vprep(const h16* __restrict__ Vh, h16* __restrict__ Vth,
                        float* __restrict__ Vtail) {
  int bh = blockIdx.x;
  int d = threadIdx.x;
  const h16* Vb = Vh + (size_t)bh * NSEQ * DHEAD;
  h16* Vtb = Vth + (size_t)bh * DHEAD * KPAD;
  float s = 0.f;
  for (int j = 0; j < NSEQ; ++j) {
    h16 v = Vb[(size_t)j * DHEAD + d];
    if (j < KPAD) Vtb[(size_t)d * KPAD + j] = v;
    if (j >= NDIAG) s += (float)v;
  }
  Vtail[bh * DHEAD + d] = s;
}

// ---------------------------------------------------------------------------
// K2: attention. One wave per 16-row tile per (b,h).
//  Phase 1: band[r][c] = Q[n0+r] . K[n0-128+c], c in [0,272), via 17 WMMA tiles
//  Phase 2: faithful softmax: row max includes 0; Z includes 1791*exp(-mx)
//  Phase 3: out = A(16x288) @ Vhead(288x64) + coef * Vtail   (9 WMMA k-steps)
// grid = (128, 24), block = 32
// ---------------------------------------------------------------------------
__global__ __launch_bounds__(32) void k_attn(const h16* __restrict__ Qh,
                                             const h16* __restrict__ Kh,
                                             const h16* __restrict__ Vth,
                                             const float* __restrict__ Vtail,
                                             h16* __restrict__ attn_h) {
  int n0 = blockIdx.x * 16;
  int bh = blockIdx.y;
  int lane = threadIdx.x & 31;
  int lo = lane & 15, hi = lane >> 4;

  const h16* Qb = Qh + (size_t)bh * NSEQ * DHEAD;
  const h16* Kb = Kh + (size_t)bh * NSEQ * DHEAD;
  const h16* Vtb = Vth + (size_t)bh * DHEAD * KPAD;

  __shared__ float band[16][280];   // 272 used, padded
  __shared__ __align__(16) h16 Ash[16][KPAD];
  __shared__ float coef[16];

  // Q tile fragments (16 x 64 -> two 16x32 A-frags)
  v16h q0 = load_a16x32(Qb + (size_t)n0 * DHEAD, DHEAD, 0);
  v16h q1 = load_a16x32(Qb + (size_t)n0 * DHEAD, DHEAD, 32);

  // Phase 1: P tiles.  B-frag "columns" are K rows m0+lo; K row-major (m,d) so
  // per-lane data is contiguous in d (same as transposed-B loader semantics).
  for (int ct = 0; ct < 17; ++ct) {
    int m0 = n0 - 128 + ct * 16;
    int mcol = m0 + lo;
    bool ok = (mcol >= 0) && (mcol < NSEQ);
    v16h b0 = {}, b1 = {};
    if (ok) {
      const h16* kp = Kb + (size_t)mcol * DHEAD + hi * 16;
#pragma unroll
      for (int e = 0; e < 16; ++e) b0[e] = kp[e];
#pragma unroll
      for (int e = 0; e < 16; ++e) b1[e] = kp[32 + e];
    }
    v8f acc = {};
    acc = wmma_f16(q0, b0, acc);
    acc = wmma_f16(q1, b1, acc);
#pragma unroll
    for (int v = 0; v < 8; ++v) band[v + hi * 8][ct * 16 + lo] = acc[v];
  }
  __syncthreads();

  // Phase 2: per-row softmax stats (16 lanes, one row each)
  if (lane < 16) {
    int r = lane;
    float mx = 0.f;  // zero columns 257..2047 of the score buffer set the max floor
    for (int c = 0; c < 272; ++c) {
      int j = c - r;
      if ((unsigned)j > 256u) continue;
      int m = n0 - 128 + c;
      float val = (m >= 0 && m < NSEQ) ? band[r][c] : 0.f;
      mx = fmaxf(mx, val);
    }
    float s = (float)TAILN * __expf(-mx);
    for (int c = 0; c < 272; ++c) {
      int j = c - r;
      if ((unsigned)j > 256u) continue;
      int m = n0 - 128 + c;
      float val = (m >= 0 && m < NSEQ) ? band[r][c] : 0.f;
      s += __expf(val - mx);
    }
    float inv = 1.f / s;
    for (int c = 0; c < 272; ++c) {
      int j = c - r;
      if ((unsigned)j > 256u) continue;
      int m = n0 - 128 + c;
      float val = (m >= 0 && m < NSEQ) ? band[r][c] : 0.f;
      Ash[r][j] = (h16)(__expf(val - mx) * inv);
    }
    for (int j = NDIAG; j < KPAD; ++j) Ash[r][j] = (h16)0.f;
    coef[r] = __expf(-mx) * inv;
  }
  __syncthreads();

  // Phase 3: A @ V_head (16x288 @ 288x64), 4 col-tiles, 9 k-steps
  v8f oacc[4];
#pragma unroll
  for (int t = 0; t < 4; ++t) {
    v8f z = {};
    oacc[t] = z;
  }
  for (int k0 = 0; k0 < KPAD; k0 += 32) {
    v16h aA = load_a16x32(&Ash[0][0], KPAD, k0);
#pragma unroll
    for (int t = 0; t < 4; ++t) {
      v16h bV = load_b32x16T(Vtb, KPAD, t * 16, k0);
      oacc[t] = wmma_f16(aA, bV, oacc[t]);
    }
  }

  int b = bh / NH, h = bh % NH;
#pragma unroll
  for (int t = 0; t < 4; ++t) {
#pragma unroll
    for (int v = 0; v < 8; ++v) {
      int r = v + hi * 8;
      int d = t * 16 + lo;
      float val = oacc[t][v] + coef[r] * Vtail[bh * DHEAD + d];
      attn_h[((size_t)(b * NSEQ + n0 + r)) * EMBED + h * DHEAD + d] = (h16)val;
    }
  }
}

// ---------------------------------------------------------------------------
// Host-side launcher
// ---------------------------------------------------------------------------
extern "C" void kernel_launch(void* const* d_in, const int* in_sizes, int n_in,
                              void* d_out, int out_size, void* d_ws, size_t ws_size,
                              hipStream_t stream) {
  const float* x  = (const float*)d_in[0];
  const float* Wq = (const float*)d_in[1];
  const float* bq = (const float*)d_in[2];
  const float* Wk = (const float*)d_in[3];
  const float* bk = (const float*)d_in[4];
  const float* Wv = (const float*)d_in[5];
  const float* bv = (const float*)d_in[6];
  const float* Wo = (const float*)d_in[7];
  const float* bo = (const float*)d_in[8];
  float* out = (float*)d_out;

  // workspace layout
  char* ws = (char*)d_ws;
  size_t off = 0;
  h16* xh  = (h16*)(ws + off); off += (size_t)MROWS * EMBED * sizeof(h16);
  h16* WqT = (h16*)(ws + off); off += (size_t)EMBED * EMBED * sizeof(h16);
  h16* WkT = (h16*)(ws + off); off += (size_t)EMBED * EMBED * sizeof(h16);
  h16* WvT = (h16*)(ws + off); off += (size_t)EMBED * EMBED * sizeof(h16);
  h16* WoT = (h16*)(ws + off); off += (size_t)EMBED * EMBED * sizeof(h16);
  h16* Qh  = (h16*)(ws + off); off += (size_t)BB * NH * NSEQ * DHEAD * sizeof(h16);
  h16* Kh  = (h16*)(ws + off); off += (size_t)BB * NH * NSEQ * DHEAD * sizeof(h16);
  h16* Vh  = (h16*)(ws + off); off += (size_t)BB * NH * NSEQ * DHEAD * sizeof(h16);
  h16* Vth = (h16*)(ws + off); off += (size_t)BB * NH * DHEAD * KPAD * sizeof(h16);
  float* Vtail = (float*)(ws + off); off += (size_t)BB * NH * DHEAD * sizeof(float);
  h16* attn_h = (h16*)(ws + off); off += (size_t)MROWS * EMBED * sizeof(h16);

  // K0: conversions
  {
    int n = MROWS * EMBED;
    k_cvt_f16<<<(n + 255) / 256, 256, 0, stream>>>(x, xh, n);
    int nw = EMBED * EMBED;
    k_cvt_T<<<(nw + 255) / 256, 256, 0, stream>>>(Wq, WqT, EMBED, EMBED);
    k_cvt_T<<<(nw + 255) / 256, 256, 0, stream>>>(Wk, WkT, EMBED, EMBED);
    k_cvt_T<<<(nw + 255) / 256, 256, 0, stream>>>(Wv, WvT, EMBED, EMBED);
    k_cvt_T<<<(nw + 255) / 256, 256, 0, stream>>>(Wo, WoT, EMBED, EMBED);
  }

  // K1: QKV projections (WMMA GEMM with async-LDS A staging)
  dim3 ggrid(MROWS / 16, EMBED / 256);
  k_gemm16<<<ggrid, 128, 0, stream>>>(xh, WqT, bq, (void*)Qh, 0);
  k_gemm16<<<ggrid, 128, 0, stream>>>(xh, WkT, bk, (void*)Kh, 0);
  k_gemm16<<<ggrid, 128, 0, stream>>>(xh, WvT, bv, (void*)Vh, 0);

  // K3: V head transpose + tail sums
  k_vprep<<<BB * NH, DHEAD, 0, stream>>>(Vh, Vth, Vtail);

  // K2: banded attention with faithful dense-softmax semantics
  dim3 agrid(NSEQ / 16, BB * NH);
  k_attn<<<agrid, 32, 0, stream>>>(Qh, Kh, Vth, Vtail, attn_h);

  // K4: output projection -> f32
  k_gemm16<<<ggrid, 128, 0, stream>>>(attn_h, WoT, bo, (void*)out, 1);
}
